// MultiQueryAttention_6571299963364
// MI455X (gfx1250) — compile-verified, matrix operand order checked by reading, not run on hardware
//
#include <hip/hip_runtime.h>
#include <hip/hip_bf16.h>

// ---------- types ----------
typedef __attribute__((ext_vector_type(16))) __bf16        v16bf;
typedef __attribute__((ext_vector_type(8)))  float         v8f;
typedef __attribute__((ext_vector_type(4)))  unsigned int  u32x4;

union FragBF { v16bf v; u32x4 u[2]; };

__device__ __forceinline__ unsigned short f2bf(float f) {
  unsigned int u = __float_as_uint(f);
  u += 0x7FFFu + ((u >> 16) & 1u);           // round-to-nearest-even
  return (unsigned short)(u >> 16);
}

__device__ __forceinline__ u32x4 ld16g(const unsigned short* p) {
  return *reinterpret_cast<const u32x4*>(p); // global_load_b128
}

__device__ __forceinline__ v8f wmma_bf16(const FragBF& a, const FragBF& b, v8f c) {
  return __builtin_amdgcn_wmma_f32_16x16x32_bf16(false, a.v, false, b.v,
                                                 (short)0, c, false, false);
}

// ---------- elementwise converters ----------
__global__ void k_f32_to_bf16(const float* __restrict__ in,
                              unsigned short* __restrict__ out, int n) {
  int i = blockIdx.x * blockDim.x + threadIdx.x;
  if (i < n) out[i] = f2bf(in[i]);
}

// W [K,N] fp32 row-major -> Wt [N,K] bf16 row-major
__global__ void k_transpose_bf16(const float* __restrict__ W,
                                 unsigned short* __restrict__ Wt, int K, int N) {
  int i = blockIdx.x * blockDim.x + threadIdx.x;
  if (i >= K * N) return;
  int k = i / N, n = i % N;
  Wt[(size_t)n * K + k] = f2bf(W[i]);
}

// ---------- bf16 GEMM: C[M,N] = A[M,K] * Bt[N,K]^T ----------
// One wave computes a 32x64 tile: 2 A-fragments x 4 B-fragments = 8 WMMAs per
// K-step of 32 (~22 FLOP/byte from cache). Waves in a block share the same
// 64-wide N-block (B-fragment reuse in WGP$) and walk consecutive M-tiles.
// mode 0: bf16 row-major store.  mode 1: bf16 transposed per-batch store
// (out[(row/2048)*N + col][row%2048], for V^T).  mode 2: fp32 row-major store.
__global__ __launch_bounds__(256) void k_gemm_bf16(
    const unsigned short* __restrict__ A,
    const unsigned short* __restrict__ Bt,
    void* __restrict__ C, int M, int N, int K, int mode) {
  const int lane = threadIdx.x & 31;
  const int hf   = lane >> 4;        // half-wave selector
  const int n16  = lane & 15;
  const int wid  = blockIdx.x * (blockDim.x >> 5) + (threadIdx.x >> 5);
  const int mTiles = M >> 5;                    // 32-row M tiles
  const int total  = mTiles * (N >> 6);
  if (wid >= total) return;
  const int mt = wid % mTiles;                  // consecutive waves: same nb
  const int nb = (wid / mTiles) << 6;

  const unsigned short* arow0 = A  + (size_t)(mt * 32 + n16) * K;
  const unsigned short* arow1 = arow0 + (size_t)16 * K;
  const unsigned short* brow0 = Bt + (size_t)(nb + n16) * K;
  const size_t bstride = (size_t)16 * K;

  v8f c00 = {}, c01 = {}, c02 = {}, c03 = {};
  v8f c10 = {}, c11 = {}, c12 = {}, c13 = {};
  for (int k0 = 0; k0 < K; k0 += 32) {
    FragBF a0, a1;
    a0.u[0] = ld16g(arow0 + k0 + 8 * hf);
    a0.u[1] = ld16g(arow0 + k0 + 16 + 8 * hf);
    a1.u[0] = ld16g(arow1 + k0 + 8 * hf);
    a1.u[1] = ld16g(arow1 + k0 + 16 + 8 * hf);
    __builtin_prefetch(arow0 + k0 + 128, 0, 1);   // global_prefetch_b8
    __builtin_prefetch(arow1 + k0 + 128, 0, 1);
    FragBF b0, b1, b2, b3;
    b0.u[0] = ld16g(brow0 + k0 + 8 * hf);
    b0.u[1] = ld16g(brow0 + k0 + 16 + 8 * hf);
    b1.u[0] = ld16g(brow0 + bstride + k0 + 8 * hf);
    b1.u[1] = ld16g(brow0 + bstride + k0 + 16 + 8 * hf);
    b2.u[0] = ld16g(brow0 + 2 * bstride + k0 + 8 * hf);
    b2.u[1] = ld16g(brow0 + 2 * bstride + k0 + 16 + 8 * hf);
    b3.u[0] = ld16g(brow0 + 3 * bstride + k0 + 8 * hf);
    b3.u[1] = ld16g(brow0 + 3 * bstride + k0 + 16 + 8 * hf);
    c00 = wmma_bf16(a0, b0, c00);
    c01 = wmma_bf16(a0, b1, c01);
    c02 = wmma_bf16(a0, b2, c02);
    c03 = wmma_bf16(a0, b3, c03);
    c10 = wmma_bf16(a1, b0, c10);
    c11 = wmma_bf16(a1, b1, c11);
    c12 = wmma_bf16(a1, b2, c12);
    c13 = wmma_bf16(a1, b3, c13);
  }

#pragma unroll
  for (int g = 0; g < 2; ++g) {
    const v8f& d0 = g ? c10 : c00;
    const v8f& d1 = g ? c11 : c01;
    const v8f& d2 = g ? c12 : c02;
    const v8f& d3 = g ? c13 : c03;
#pragma unroll
    for (int r = 0; r < 8; ++r) {
      const int row = mt * 32 + g * 16 + r + 8 * hf;
      const float v0 = d0[r], v1 = d1[r], v2 = d2[r], v3 = d3[r];
      if (mode == 0) {
        unsigned short* o = (unsigned short*)C + (size_t)row * N + nb + n16;
        o[0] = f2bf(v0); o[16] = f2bf(v1); o[32] = f2bf(v2); o[48] = f2bf(v3);
      } else if (mode == 1) {
        const int bb = row >> 11, s = row & 2047;     // S = 2048
        unsigned short* o =
            (unsigned short*)C + ((size_t)bb * N + nb + n16) * 2048 + s;
        o[0]           = f2bf(v0);
        o[16u * 2048u] = f2bf(v1);
        o[32u * 2048u] = f2bf(v2);
        o[48u * 2048u] = f2bf(v3);
      } else {
        float* o = (float*)C + (size_t)row * N + nb + n16;
        o[0] = v0; o[16] = v1; o[32] = v2; o[48] = v3;
      }
    }
  }
}

// ---------- flash attention (MQA): one wave = 16 queries of one (b,h) ----------
// Qb  [B*S, 1024] bf16 (head h at cols h*64..h*64+63)
// Kb  [B*S, 64]   bf16
// Vt  [B, 64, S]  bf16 (transposed V)
// AO  [B*S, 1024] bf16 (attention output, pre-Wo)
__global__ __launch_bounds__(256) void k_attn(
    const unsigned short* __restrict__ Qb,
    const unsigned short* __restrict__ Kb,
    const unsigned short* __restrict__ Vt,
    unsigned short* __restrict__ AO) {
  __shared__ unsigned short Plds[8][16 * 32];   // 1KB private P-staging per wave
  const int lane = threadIdx.x & 31;
  const int hf   = lane >> 4;
  const int n16  = lane & 15;
  const int wave = threadIdx.x >> 5;
  const int wid  = blockIdx.x * 8 + wave;
  const int qt = wid & 127;           // S/16 = 128 query tiles
  const int h  = (wid >> 7) & 15;
  const int b  = wid >> 11;
  if (b >= 4) return;

  // Q A-fragments (hd=64 -> two K=32 fragments), reused for all 2048 keys
  const unsigned short* qrow =
      Qb + (size_t)(b * 2048 + qt * 16 + n16) * 1024 + h * 64;
  FragBF qa0, qa1;
  qa0.u[0] = ld16g(qrow + 8 * hf);
  qa0.u[1] = ld16g(qrow + 16 + 8 * hf);
  qa1.u[0] = ld16g(qrow + 32 + 8 * hf);
  qa1.u[1] = ld16g(qrow + 48 + 8 * hf);

  float m[8], l[8];
#pragma unroll
  for (int r = 0; r < 8; ++r) { m[r] = -3.0e38f; l[r] = 0.f; }
  v8f o0 = {}, o1 = {}, o2 = {}, o3 = {};

  unsigned short* P = Plds[wave];
  const unsigned short* kbase = Kb + (size_t)(b * 2048 + n16) * 64;
  const unsigned short* vbase = Vt + (size_t)b * 64 * 2048;

  for (int kb = 0; kb < 2048; kb += 32) {
    // --- scores: S = Q * K^T, 32 keys per iteration (two 16-key B tiles) ---
    const unsigned short* kr0 = kbase + (size_t)kb * 64;
    const unsigned short* kr1 = kr0 + 16 * 64;
    FragBF k00, k01, k10, k11;
    k00.u[0] = ld16g(kr0 + 8 * hf);      k00.u[1] = ld16g(kr0 + 16 + 8 * hf);
    k01.u[0] = ld16g(kr0 + 32 + 8 * hf); k01.u[1] = ld16g(kr0 + 48 + 8 * hf);
    k10.u[0] = ld16g(kr1 + 8 * hf);      k10.u[1] = ld16g(kr1 + 16 + 8 * hf);
    k11.u[0] = ld16g(kr1 + 32 + 8 * hf); k11.u[1] = ld16g(kr1 + 48 + 8 * hf);

    v8f s0 = {}, s1 = {};
    s0 = wmma_bf16(qa0, k00, s0);
    s0 = wmma_bf16(qa1, k01, s0);
    s1 = wmma_bf16(qa0, k10, s1);
    s1 = wmma_bf16(qa1, k11, s1);

    // --- online softmax: rows live in fixed 16-lane halves -> xor 1/2/4/8 ---
#pragma unroll
    for (int r = 0; r < 8; ++r) {
      float a0 = s0[r] * 0.125f;           // 1/sqrt(64)
      float a1 = s1[r] * 0.125f;
      float mv = fmaxf(a0, a1);
      mv = fmaxf(mv, __shfl_xor(mv, 1, 32));
      mv = fmaxf(mv, __shfl_xor(mv, 2, 32));
      mv = fmaxf(mv, __shfl_xor(mv, 4, 32));
      mv = fmaxf(mv, __shfl_xor(mv, 8, 32));
      float nm    = fmaxf(m[r], mv);
      float alpha = __expf(m[r] - nm);
      float p0 = __expf(a0 - nm);
      float p1 = __expf(a1 - nm);
      float rs = p0 + p1;
      rs += __shfl_xor(rs, 1, 32);
      rs += __shfl_xor(rs, 2, 32);
      rs += __shfl_xor(rs, 4, 32);
      rs += __shfl_xor(rs, 8, 32);
      l[r] = l[r] * alpha + rs;
      m[r] = nm;
      o0[r] *= alpha; o1[r] *= alpha; o2[r] *= alpha; o3[r] *= alpha;
      const int q = r + 8 * hf;            // C-layout row -> P row
      P[q * 32 + n16]      = f2bf(p0);
      P[q * 32 + 16 + n16] = f2bf(p1);
    }
    asm volatile("s_wait_dscnt 0" ::: "memory");   // same-wave LDS store->load

    // re-read P in A-fragment layout (16x32, K=keys)
    FragBF pa;
    pa.u[0] = *reinterpret_cast<const u32x4*>(P + n16 * 32 + 8 * hf);
    pa.u[1] = *reinterpret_cast<const u32x4*>(P + n16 * 32 + 16 + 8 * hf);

    // --- O += P * V : B columns = head dims, contraction = keys (Vt rows) ---
    const unsigned short* vr = vbase + (size_t)n16 * 2048 + kb;
    FragBF bv0, bv1, bv2, bv3;
    bv0.u[0] = ld16g(vr + 8 * hf);             bv0.u[1] = ld16g(vr + 16 + 8 * hf);
    bv1.u[0] = ld16g(vr + 16 * 2048 + 8 * hf); bv1.u[1] = ld16g(vr + 16 * 2048 + 16 + 8 * hf);
    bv2.u[0] = ld16g(vr + 32 * 2048 + 8 * hf); bv2.u[1] = ld16g(vr + 32 * 2048 + 16 + 8 * hf);
    bv3.u[0] = ld16g(vr + 48 * 2048 + 8 * hf); bv3.u[1] = ld16g(vr + 48 * 2048 + 16 + 8 * hf);
    o0 = wmma_bf16(pa, bv0, o0);
    o1 = wmma_bf16(pa, bv1, o1);
    o2 = wmma_bf16(pa, bv2, o2);
    o3 = wmma_bf16(pa, bv3, o3);
  }

#pragma unroll
  for (int r = 0; r < 8; ++r) {
    const float inv = 1.0f / l[r];
    const int row = b * 2048 + qt * 16 + r + 8 * hf;
    unsigned short* o = AO + (size_t)row * 1024 + h * 64 + n16;
    o[0]  = f2bf(o0[r] * inv);
    o[16] = f2bf(o1[r] * inv);
    o[32] = f2bf(o2[r] * inv);
    o[48] = f2bf(o3[r] * inv);
  }
}

// ---------- launch ----------
extern "C" void kernel_launch(void* const* d_in, const int* in_sizes, int n_in,
                              void* d_out, int out_size, void* d_ws, size_t ws_size,
                              hipStream_t stream) {
  const float* x  = (const float*)d_in[0];
  const float* Wq = (const float*)d_in[1];
  const float* Wk = (const float*)d_in[2];
  const float* Wv = (const float*)d_in[3];
  const float* Wo = (const float*)d_in[4];

  constexpr int B = 4, S = 2048, D = 1024, HD = 64;
  constexpr int M = B * S;   // 8192

  char* ws = (char*)d_ws;
  size_t off = 0;
  auto alloc = [&](size_t bytes) {
    void* p = ws + off;
    off += (bytes + 255) & ~(size_t)255;
    return p;
  };
  unsigned short* xb  = (unsigned short*)alloc((size_t)M * D * 2);
  unsigned short* wqT = (unsigned short*)alloc((size_t)D * D * 2);
  unsigned short* wkT = (unsigned short*)alloc((size_t)HD * D * 2);
  unsigned short* wvT = (unsigned short*)alloc((size_t)HD * D * 2);
  unsigned short* woT = (unsigned short*)alloc((size_t)D * D * 2);
  unsigned short* Qb  = (unsigned short*)alloc((size_t)M * D * 2);
  unsigned short* Kb  = (unsigned short*)alloc((size_t)M * HD * 2);
  unsigned short* Vt  = (unsigned short*)alloc((size_t)M * HD * 2);
  unsigned short* AO  = (unsigned short*)alloc((size_t)M * D * 2);

  // 1) precision conversion (fp32 -> bf16), weights transposed to [N,K]
  k_f32_to_bf16<<<(M * D + 255) / 256, 256, 0, stream>>>(x, xb, M * D);
  k_transpose_bf16<<<(D * D + 255) / 256, 256, 0, stream>>>(Wq, wqT, D, D);
  k_transpose_bf16<<<(D * HD + 255) / 256, 256, 0, stream>>>(Wk, wkT, D, HD);
  k_transpose_bf16<<<(D * HD + 255) / 256, 256, 0, stream>>>(Wv, wvT, D, HD);
  k_transpose_bf16<<<(D * D + 255) / 256, 256, 0, stream>>>(Wo, woT, D, D);

  // 2) projections: Q [M,D], K [M,64], V stored transposed per batch [B,64,S]
  {
    int waves = (M >> 5) * (D >> 6);       // 4096
    k_gemm_bf16<<<waves / 8, 256, 0, stream>>>(xb, wqT, Qb, M, D, D, 0);
  }
  {
    int waves = (M >> 5) * (HD >> 6);      // 256
    k_gemm_bf16<<<waves / 8, 256, 0, stream>>>(xb, wkT, Kb, M, HD, D, 0);
    k_gemm_bf16<<<waves / 8, 256, 0, stream>>>(xb, wvT, Vt, M, HD, D, 1);
  }

  // 3) flash attention: 8192 waves (B*H*S/16), 8 waves/block
  k_attn<<<1024, 256, 0, stream>>>(Qb, Kb, Vt, AO);

  // 4) output projection -> fp32 d_out
  {
    int waves = (M >> 5) * (D >> 6);       // 4096
    k_gemm_bf16<<<waves / 8, 256, 0, stream>>>(AO, woT, d_out, M, D, D, 2);
  }
}
